// ZS_MKAN_72121090834671
// MI455X (gfx1250) — matrix operand order repaired and checked by reading, not compiled, not run on hardware
//
#include <hip/hip_runtime.h>
#include <hip/hip_bf16.h>

// ---------------------------------------------------------------------------
// KAN conv net on gfx1250 (MI455X), v2: tap-deduplicated.
//   stage1: KANConv 3x3 grid3, 3ch->6ch      (WMMA, K=96,  3 chunks)
//   stage2: KANConv 3x3 grid3 folded w/ 1x1  (WMMA, K=448, 14 chunks, 3 outs)
//   stage4: pointwise KAN grid5              (VALU)
// Unique tap vectors [silu, B0..B5, 0] are computed once per block into an
// LDS table (each input value feeds 9 output pixels); feature rows are then
// assembled with aligned ds_load_b128/ds_store_b128 copies.
// ---------------------------------------------------------------------------

typedef __attribute__((ext_vector_type(16))) _Float16 v16h;
typedef __attribute__((ext_vector_type(8)))  _Float16 v8h;
typedef __attribute__((ext_vector_type(8)))  float    v8f;

#define HW 65536   // 256*256 pixels per plane

__device__ __forceinline__ float siluf(float x) {
    return x / (1.0f + __expf(-x));
}

// Unique tap vector for grid_size=3 (uniform knots -3..3 step 2/3):
// [silu(x), B0(x)..B5(x), 0].  Only 4 bases nonzero; rest forced to 0.
__device__ __forceinline__ v8h tapvec_g3(float x) {
    float s = siluf(x);
    float u = (x + 3.0f) * 1.5f;            // span coordinate, valid [0,9)
    float w0 = 0.f, w1 = 0.f, w2 = 0.f, w3 = 0.f;
    int   c0 = 99;                           // sentinel: no basis hit
    if (u >= 0.0f && u < 9.0f) {
        int   j  = (int)u;
        float t  = u - (float)j;
        float omt = 1.0f - t;
        float t2 = t * t, t3 = t2 * t;
        w0 = omt * omt * omt * (1.0f / 6.0f);
        w1 = (3.0f * t3 - 6.0f * t2 + 4.0f) * (1.0f / 6.0f);
        w2 = (-3.0f * t3 + 3.0f * t2 + 3.0f * t + 1.0f) * (1.0f / 6.0f);
        w3 = t3 * (1.0f / 6.0f);
        c0 = j - 3;
    }
    v8h o;
    o[0] = (_Float16)s;
#pragma unroll
    for (int c = 0; c < 6; ++c) {
        float wv = 0.f;
        if      (c == c0)     wv = w0;
        else if (c == c0 + 1) wv = w1;
        else if (c == c0 + 2) wv = w2;
        else if (c == c0 + 3) wv = w3;
        o[1 + c] = (_Float16)wv;
    }
    o[7] = (_Float16)0.f;                    // pad slot: must be 0 (WMMA reads it)
    return o;
}

// Pointwise KAN, grid_size=5 (knots -2.2..2.2 step 0.4, 8 coefficients).
__device__ __forceinline__ float kan6(float v, float bw6, const float* sw6) {
    float r = bw6 * siluf(v);
    float u = (v + 2.2f) * 2.5f;            // valid [0,11)
    if (u >= 0.0f && u < 11.0f) {
        int   j  = (int)u;
        float t  = u - (float)j;
        float omt = 1.0f - t;
        float t2 = t * t, t3 = t2 * t;
        float w0 = omt * omt * omt * (1.0f / 6.0f);
        float w1 = (3.0f * t3 - 6.0f * t2 + 4.0f) * (1.0f / 6.0f);
        float w2 = (-3.0f * t3 + 3.0f * t2 + 3.0f * t + 1.0f) * (1.0f / 6.0f);
        float w3 = t3 * (1.0f / 6.0f);
        int c0 = j - 3;
        if (c0     >= 0 && c0     < 8) r += sw6[c0]     * w0;
        if (c0 + 1 >= 0 && c0 + 1 < 8) r += sw6[c0 + 1] * w1;
        if (c0 + 2 >= 0 && c0 + 2 < 8) r += sw6[c0 + 2] * w2;
        if (c0 + 3 >= 0 && c0 + 3 < 8) r += sw6[c0 + 3] * w3;
    }
    return r;
}

// B fragment (32x16 f16) for K-chunk q from f32 table Wt[K][16].
// contiguous k order used here (not AMD table order): lane<16 -> col=lane, K=q*32+0..15; lane>=16 -> K=q*32+16..31.
__device__ __forceinline__ v16h load_bfrag_global(const float* Wt, int q, int lane) {
    int n  = lane & 15;
    int kb = (lane < 16) ? 0 : 16;
    v16h b;
#pragma unroll
    for (int e = 0; e < 16; ++e)
        b[e] = (_Float16)Wt[(q * 32 + kb + e) * 16 + n];
    return b;
}

// A fragment (16x32 f16) from an LDS feature row (rowp points at K-chunk).
// contiguous k order (same as B): lane<16 -> M=lane,   K = 0..15
//                  lane>=16 -> M=lane-16, K = 16..31
__device__ __forceinline__ v16h load_afrag(const _Float16* rowp, int lane) {
    int off = (lane < 16) ? 0 : 16;
    v8h lo = *(const v8h*)(rowp + off);
    v8h hi = *(const v8h*)(rowp + 8 + off);
    v16h a;
#pragma unroll
    for (int e = 0; e < 8; ++e) { a[e] = lo[e]; a[8 + e] = hi[e]; }
    return a;
}

// ---------------------------------------------------------------------------
// Kernel 0: weight tables.  Feature layout is 8 halfs per tap:
//   f = tap*8 + s, s=0 silu weight, s=1..6 spline coefs (scaler folded),
//   s=7 pad (zero).  Wt1: 96x16 (2 cols used).  Wt2: 448x16 (3 cols used,
//   1x1 conv_w pre-contracted over the 12 stage2 channels).
// ---------------------------------------------------------------------------
__global__ void kan_tables(const float* __restrict__ bw1, const float* __restrict__ sw1,
                           const float* __restrict__ sc1, const float* __restrict__ bw2,
                           const float* __restrict__ sw2, const float* __restrict__ sc2,
                           const float* __restrict__ cw,
                           float* __restrict__ Wt1, float* __restrict__ Wt2) {
    for (int idx = threadIdx.x; idx < 96 * 16; idx += blockDim.x) {
        int f = idx >> 4, n = idx & 15;
        int i = f >> 3, s = f & 7;
        float v = 0.0f;
        if (n < 2 && f < 72 && s < 7) {
            v = (s == 0) ? bw1[n * 9 + i]
                         : sw1[(n * 9 + i) * 6 + (s - 1)] * sc1[n * 9 + i];
        }
        Wt1[f * 16 + n] = v;
    }
    for (int idx = threadIdx.x; idx < 448 * 16; idx += blockDim.x) {
        int f = idx >> 4, n = idx & 15;
        float v = 0.0f;
        if (n < 3 && f < 432 && (f & 7) < 7) {
            int c6 = f / 72, r = f % 72;
            int i = r >> 3, s = r & 7;
#pragma unroll
            for (int k = 0; k < 2; ++k) {
                float w   = cw[n * 12 + (c6 * 2 + k)];       // conv_w[o][12]
                float phi = (s == 0) ? bw2[k * 9 + i]
                                     : sw2[(k * 9 + i) * 6 + (s - 1)] * sc2[k * 9 + i];
                v += w * phi;
            }
        }
        Wt2[f * 16 + n] = v;
    }
}

// ---------------------------------------------------------------------------
// Kernel 1: stage1.  128 threads = 4 waves, block covers 128 pixels of one
// image row of one plane.  Unique taps (3 rows x 130 cols) computed once into
// LDS, then each wave assembles 32 feature rows (K=96) and does 2 tiles x 3
// WMMA chunks -> 2 output channels.
// ---------------------------------------------------------------------------
__global__ __launch_bounds__(128) void kan_stage1(const float* __restrict__ x,
                                                  const float* __restrict__ Wt1,
                                                  float* __restrict__ T1) {
    __shared__ __align__(16) _Float16 feat[4][32][104];   // K=96, stride 104
    __shared__ __align__(16) _Float16 tt[3][132][8];      // 130 cols used

    const int lane   = threadIdx.x & 31;
    const int wv     = threadIdx.x >> 5;
    const int plane  = blockIdx.x >> 9;                   // b*3+c, 512 blk/plane
    const int blkpix = (blockIdx.x & 511) << 7;           // 128 px per block
    const int b = plane / 3, c = plane % 3;
    const int y   = blkpix >> 8;
    const int xcb = blkpix & 255;
    const float* in = x + (size_t)plane * HW;

    // unique tap table: rows y-1..y+1, cols xcb-1..xcb+128 (zero-padded border)
    for (int idx = threadIdx.x; idx < 3 * 130; idx += 128) {
        int r = idx / 130, col = idx % 130;
        int gy = y + r - 1, gx = xcb + col - 1;
        float v = (gy >= 0 && gy < 256 && gx >= 0 && gx < 256) ? in[gy * 256 + gx] : 0.0f;
        *(v8h*)&tt[r][col][0] = tapvec_g3(v);
    }
    __syncthreads();

    v16h bf0 = load_bfrag_global(Wt1, 0, lane);
    v16h bf1 = load_bfrag_global(Wt1, 1, lane);
    v16h bf2 = load_bfrag_global(Wt1, 2, lane);

    // assemble this lane's feature row from the tap table
    _Float16* row = &feat[wv][lane][0];
    const int px = (wv << 5) + lane;                       // pixel within block
    {
        v8h z = {};
        *(v8h*)(row + 72) = z;  *(v8h*)(row + 80) = z;  *(v8h*)(row + 88) = z;
    }
#pragma unroll
    for (int i = 0; i < 9; ++i) {
        int dy = i / 3, dx = i % 3;                        // 0..2
        v8h tv = *(const v8h*)&tt[dy][px + dx][0];
        *(v8h*)(row + i * 8) = tv;
    }
    asm volatile("s_wait_dscnt 0" ::: "memory");           // wave's stores visible

#pragma unroll
    for (int t = 0; t < 2; ++t) {
        const _Float16* tr = &feat[wv][t * 16 + (lane & 15)][0];
        v8f acc = {0.f, 0.f, 0.f, 0.f, 0.f, 0.f, 0.f, 0.f};
        v16h a0 = load_afrag(tr +  0, lane);
        acc = __builtin_amdgcn_wmma_f32_16x16x32_f16(false, a0, false, bf0,
                                                     (short)0, acc, false, false);
        v16h a1 = load_afrag(tr + 32, lane);
        acc = __builtin_amdgcn_wmma_f32_16x16x32_f16(false, a1, false, bf1,
                                                     (short)0, acc, false, false);
        v16h a2 = load_afrag(tr + 64, lane);
        acc = __builtin_amdgcn_wmma_f32_16x16x32_f16(false, a2, false, bf2,
                                                     (short)0, acc, false, false);
        if ((lane & 15) < 2) {
            int k     = lane & 15;
            int mbase = (lane < 16) ? 0 : 8;
            float* op = T1 + ((size_t)(b * 6 + c * 2 + k)) * HW
                          + blkpix + (wv << 5) + t * 16 + mbase;
#pragma unroll
            for (int r = 0; r < 8; ++r) op[r] = acc[r];
        }
    }
}

// ---------------------------------------------------------------------------
// Kernel 2: stage2 (conv_w folded, 3 outputs) + bias + stage4 KAN.
// 64 threads = 2 waves, block covers 32 pixels of one row.  Unique taps
// (6ch x 3 rows x 34 cols) computed once; each wave assembles 16 feature rows
// (K=448) -> 14 accumulated WMMAs -> +bias -> pointwise KAN -> out.
// ---------------------------------------------------------------------------
__global__ __launch_bounds__(64) void kan_stage2_fused(
        const float* __restrict__ T1, const float* __restrict__ Wt2,
        const float* __restrict__ conv_b, const float* __restrict__ bw6p,
        const float* __restrict__ sw6p, const float* __restrict__ sc6p,
        float* __restrict__ out) {
    __shared__ __align__(16) _Float16 feat[2][16][456];   // K=448, stride 456
    __shared__ __align__(16) _Float16 tt[6][3][36][8];    // 34 cols used
    __shared__ __align__(32) _Float16 Bp[14][32][16];

    const int lane   = threadIdx.x & 31;
    const int wv     = threadIdx.x >> 5;
    const int batch  = blockIdx.x >> 11;                  // 2048 blocks/batch
    const int blkpix = (blockIdx.x & 2047) << 5;          // 32 px per block
    const int y   = blkpix >> 8;
    const int xcb = blkpix & 255;

    // B-fragment pack (14 chunks x 32 lanes, shared by both waves)
    for (int idx = threadIdx.x; idx < 448; idx += 64) {
        int q = idx >> 5, l = idx & 31;
        int n = l & 15, kb = (l < 16) ? 0 : 16;
#pragma unroll
        for (int e = 0; e < 16; ++e)
            Bp[q][l][e] = (_Float16)Wt2[(q * 32 + kb + e) * 16 + n];
    }
    // unique tap table: 6 channels x rows y-1..y+1 x cols xcb-1..xcb+32
    for (int idx = threadIdx.x; idx < 6 * 3 * 34; idx += 64) {
        int ch  = idx / 102;
        int rem = idx % 102;
        int r = rem / 34, col = rem % 34;
        int gy = y + r - 1, gx = xcb + col - 1;
        const float* in = T1 + ((size_t)batch * 6 + ch) * HW;
        float v = (gy >= 0 && gy < 256 && gx >= 0 && gx < 256) ? in[gy * 256 + gx] : 0.0f;
        *(v8h*)&tt[ch][r][col][0] = tapvec_g3(v);
    }
    __syncthreads();

    const float bw6 = bw6p[0];
    const float sc6 = sc6p[0];
    float sw6[8];
#pragma unroll
    for (int cc = 0; cc < 8; ++cc) sw6[cc] = sw6p[cc] * sc6;

    const int pr     = lane & 15;
    const int cstart = (lane < 16) ? 0 : 3;               // lane pair: 3 ch each
    const int px     = (wv << 4) + pr;                    // pixel within block

    // assemble features: lane covers channels cstart..cstart+2 of row pr
    _Float16* row = &feat[wv][pr][0];
    if (lane >= 16) {                                     // zero K pad 432..447
        v8h z = {};
        *(v8h*)(row + 432) = z;  *(v8h*)(row + 440) = z;
    }
#pragma unroll
    for (int cl = 0; cl < 3; ++cl) {
        int ch = cstart + cl;
#pragma unroll
        for (int i = 0; i < 9; ++i) {
            int dy = i / 3, dx = i % 3;
            v8h tv = *(const v8h*)&tt[ch][dy][px + dx][0];
            *(v8h*)(row + (ch * 9 + i) * 8) = tv;
        }
    }
    asm volatile("s_wait_dscnt 0" ::: "memory");

    v8f acc = {0.f, 0.f, 0.f, 0.f, 0.f, 0.f, 0.f, 0.f};
#pragma unroll
    for (int q = 0; q < 14; ++q) {
        v16h a  = load_afrag(&feat[wv][pr][q * 32], lane);
        v16h bb = *(const v16h*)(&Bp[q][lane][0]);
        acc = __builtin_amdgcn_wmma_f32_16x16x32_f16(false, a, false, bb,
                                                     (short)0, acc, false, false);
    }

    if ((lane & 15) < 3) {
        int o     = lane & 15;
        int mbase = (lane < 16) ? 0 : 8;
        float cb  = conv_b[o];
        float* op = out + ((size_t)batch * 3 + o) * HW + blkpix + (wv << 4) + mbase;
#pragma unroll
        for (int r = 0; r < 8; ++r) {
            float v = acc[r] + cb;              // 1x1 conv output (folded GEMM)
            op[r] = kan6(v, bw6, sw6);          // stage4 pointwise KAN
        }
    }
}

// ---------------------------------------------------------------------------
extern "C" void kernel_launch(void* const* d_in, const int* in_sizes, int n_in,
                              void* d_out, int out_size, void* d_ws, size_t ws_size,
                              hipStream_t stream) {
    const float* x   = (const float*)d_in[0];
    const float* bw1 = (const float*)d_in[1];
    const float* sw1 = (const float*)d_in[2];
    const float* sc1 = (const float*)d_in[3];
    const float* bw2 = (const float*)d_in[4];
    const float* sw2 = (const float*)d_in[5];
    const float* sc2 = (const float*)d_in[6];
    const float* cw  = (const float*)d_in[7];
    const float* cb  = (const float*)d_in[8];
    const float* bw6 = (const float*)d_in[9];
    const float* sw6 = (const float*)d_in[10];
    const float* sc6 = (const float*)d_in[11];

    float* Wt1 = (float*)d_ws;                       // 96*16 f32
    float* Wt2 = Wt1 + 1536;                         // 448*16 f32
    float* T1  = (float*)((char*)d_ws + 65536);      // (4,6,256,256) f32, 6.29MB
    float* out = (float*)d_out;

    kan_tables<<<dim3(1), dim3(256), 0, stream>>>(bw1, sw1, sc1, bw2, sw2, sc2,
                                                  cw, Wt1, Wt2);
    kan_stage1<<<dim3(6144), dim3(128), 0, stream>>>(x, Wt1, T1);
    kan_stage2_fused<<<dim3(8192), dim3(64), 0, stream>>>(T1, Wt2, cb, bw6, sw6,
                                                          sc6, out);
}
